// QEffMolmoBlock_49323404427620
// MI455X (gfx1250) — compile-verified, matrix operand order checked by reading, not run on hardware
//
#include <hip/hip_runtime.h>

#define T_LEN   2048
#define DM      2048
#define KVD     512
#define NH      16
#define KVH     4
#define HD      128

#define KROW 136   // 128 + 8 bf16 pad: 16-lane B reads land on distinct bank groups
#define VROW 72    // 64 + 8
#define PROW 72

typedef __bf16 v16bf __attribute__((ext_vector_type(16)));
typedef __bf16 v8bf  __attribute__((ext_vector_type(8)));
typedef float  v8f   __attribute__((ext_vector_type(8)));

__device__ __forceinline__ __bf16 f2bf(float f) {
    unsigned u = __float_as_uint(f);
    u += 0x7FFFu + ((u >> 16) & 1u);               // round-to-nearest-even
    unsigned short h = (unsigned short)(u >> 16);
    return __builtin_bit_cast(__bf16, h);
}

__device__ __forceinline__ v8f wmma_bf16(v16bf a, v16bf b, v8f c) {
    return __builtin_amdgcn_wmma_f32_16x16x32_bf16(false, a, false, b, (short)0, c, false, false);
}

// 32B LDS load as two 16B halves (LDS rows are 16B- but not 32B-aligned due to pad)
__device__ __forceinline__ v16bf lds_v16(const __bf16* p) {
    union { v16bf v; v8bf h[2]; } u;
    u.h[0] = *(const v8bf*)p;
    u.h[1] = *(const v8bf*)(p + 8);
    return u.v;
}

// gfx1250 async global->LDS copy, 16B per lane, tracked by ASYNCcnt
__device__ __forceinline__ void async_cp16(unsigned lds_byte, const __bf16* g) {
    asm volatile("global_load_async_to_lds_b128 %0, %1, off"
                 :: "v"(lds_byte), "v"((unsigned long long)(size_t)g)
                 : "memory");
}

// ---------------------------------------------------------------------------
// prep: RMSNorm + RoPE + head split, f32 -> bf16. One block per token.
//   Qh [NH][T][HD]  (1/sqrt(128) * log2(e) folded in -> softmax uses exp2)
//   Kh [KVH][T][HD]
//   Vt [KVH][HD][T] (transposed so PV B-operand loads are contiguous)
// ---------------------------------------------------------------------------
__global__ __launch_bounds__(256) void prep_kernel(
    const float* __restrict__ q, const float* __restrict__ k,
    const float* __restrict__ v, const int* __restrict__ pos_ids,
    const float* __restrict__ qw, const float* __restrict__ kw,
    __bf16* __restrict__ Qh, __bf16* __restrict__ Kh, __bf16* __restrict__ Vt)
{
    const int t = blockIdx.x, tid = threadIdx.x;
    __shared__ float red[256];
    __shared__ float rstat[2];
    const float* qrow = q + (size_t)t * DM;
    const float* krow = k + (size_t)t * KVD;

    float ss = 0.f;
#pragma unroll
    for (int i = 0; i < 8; ++i) { float x = qrow[tid * 8 + i]; ss += x * x; }
    red[tid] = ss; __syncthreads();
    for (int s = 128; s > 0; s >>= 1) { if (tid < s) red[tid] += red[tid + s]; __syncthreads(); }
    if (tid == 0) rstat[0] = rsqrtf(red[0] / (float)DM + 1e-5f);
    __syncthreads();

    { float x0 = krow[tid * 2], x1 = krow[tid * 2 + 1]; ss = x0 * x0 + x1 * x1; }
    red[tid] = ss; __syncthreads();
    for (int s = 128; s > 0; s >>= 1) { if (tid < s) red[tid] += red[tid + s]; __syncthreads(); }
    if (tid == 0) rstat[1] = rsqrtf(red[0] / (float)KVD + 1e-5f);
    __syncthreads();

    const float rq = rstat[0], rk = rstat[1];
    const float pos = (float)pos_ids[t];

    const int h = tid >> 4, jj = tid & 15;
#pragma unroll
    for (int p = 0; p < 4; ++p) {
        const int d = jj * 4 + p;                               // 0..63
        float ang = pos * __expf(-(float)d * (9.210340371976184f / 64.f));
        float sn, cs; __sincosf(ang, &sn, &cs);
        const int i0 = h * HD + d, i1 = i0 + 64;
        float x0 = qrow[i0] * rq * qw[i0];
        float x1 = qrow[i1] * rq * qw[i1];
        const float sc = 0.08838834764831845f * 1.4426950408889634f; // 1/sqrt(128)*log2e
        Qh[(size_t)(h * T_LEN + t) * HD + d]      = f2bf((x0 * cs - x1 * sn) * sc);
        Qh[(size_t)(h * T_LEN + t) * HD + d + 64] = f2bf((x1 * cs + x0 * sn) * sc);
    }
    if (tid < 128) {
        const int kvh = tid >> 5, kk = tid & 31;
#pragma unroll
        for (int p = 0; p < 2; ++p) {
            const int d = kk * 2 + p;
            float ang = pos * __expf(-(float)d * (9.210340371976184f / 64.f));
            float sn, cs; __sincosf(ang, &sn, &cs);
            const int i0 = kvh * HD + d, i1 = i0 + 64;
            float x0 = krow[i0] * rk * kw[i0];
            float x1 = krow[i1] * rk * kw[i1];
            Kh[(size_t)(kvh * T_LEN + t) * HD + d]      = f2bf(x0 * cs - x1 * sn);
            Kh[(size_t)(kvh * T_LEN + t) * HD + d + 64] = f2bf(x1 * cs + x0 * sn);
        }
        const float* vrow = v + (size_t)t * KVD;
#pragma unroll
        for (int p = 0; p < 4; ++p) {
            const int c = tid * 4 + p;
            Vt[(size_t)((c >> 7) * HD + (c & 127)) * T_LEN + t] = f2bf(vrow[c]);
        }
    }
}

__global__ void convw_kernel(const float* __restrict__ w, __bf16* __restrict__ wb, int n) {
    int i = blockIdx.x * blockDim.x + threadIdx.x;
    if (i < n) wb[i] = f2bf(w[i]);
}

// ---------------------------------------------------------------------------
// attn: flash attention, 64-key chunks. One block = 4 waves = one head,
// 64 consecutive queries (16/wave). All 4 waves share the K/V stream, which
// is staged into LDS once per chunk with async global->LDS (ASYNCcnt):
//   issue K(8) + V(8) per thread -> s_wait_asynccnt 8 releases score WMMAs
//   while V is still in flight   -> s_wait_asynccnt 0 gates the PV WMMAs.
// ---------------------------------------------------------------------------
__global__ __launch_bounds__(128) void attn_kernel(
    const __bf16* __restrict__ Qh, const __bf16* __restrict__ Kh,
    const __bf16* __restrict__ Vt, __bf16* __restrict__ O)
{
    __shared__ __align__(16) __bf16 KL[64 * KROW];      // 17 KB
    __shared__ __align__(16) __bf16 VL[128 * VROW];     // 18 KB
    __shared__ __align__(16) __bf16 PL[4][16 * PROW];   //  9 KB

    const int tid  = threadIdx.x;
    const int wave = tid >> 5, lane = tid & 31;
    const int half = lane >> 4, lm = lane & 15;
    const int h  = blockIdx.x >> 5;            // 16 heads
    const int qg = blockIdx.x & 31;            // 32 groups of 64 queries
    const int qt = qg * 4 + wave;
    const int qbase = qt * 16;
    const int kvh = h >> 2;
    const int nch = qg + 1;                    // same for all 4 waves

    const unsigned klb = (unsigned)(size_t)(const void*)KL;
    const unsigned vlb = (unsigned)(size_t)(const void*)VL;
    __bf16* PLw = PL[wave];

    v16bf Qa[4];
#pragma unroll
    for (int c = 0; c < 4; ++c)
        Qa[c] = *(const v16bf*)(Qh + (size_t)(h * T_LEN + qbase + lm) * HD + c * 32 + half * 16);

    v8f acc[8] = {};
    float mrow[8], lsum[8];
#pragma unroll
    for (int j = 0; j < 8; ++j) { mrow[j] = -__builtin_inff(); lsum[j] = 0.f; }

    for (int kc = 0; kc < nch; ++kc) {
        const int kb = kc * 64;
        // ---- stage K (first) then V (second) into LDS, async ----
        {
            const int r = tid >> 1, hh = tid & 1;
            const __bf16* gk = Kh + (size_t)(kvh * T_LEN + kb + r) * HD + hh * 64;
            unsigned lk = klb + (unsigned)(r * KROW + hh * 64) * 2;
#pragma unroll
            for (int j = 0; j < 8; ++j) async_cp16(lk + j * 16, gk + j * 8);
            const __bf16* gv = Vt + (size_t)(kvh * HD + tid) * T_LEN + kb;
            unsigned lv = vlb + (unsigned)(tid * VROW) * 2;
#pragma unroll
            for (int j = 0; j < 8; ++j) async_cp16(lv + j * 16, gv + j * 8);
        }
        asm volatile("s_wait_asynccnt 8" ::: "memory");   // K landed (V may be in flight)
        __syncthreads();

        // ---- scores: 4 x (16q x 16k) tiles, K-depth 128 ----
        v8f S[4];
#pragma unroll
        for (int n = 0; n < 4; ++n) {
            v8f s = {};
            const __bf16* kp = KL + (n * 16 + lm) * KROW + half * 16;
#pragma unroll
            for (int c = 0; c < 4; ++c)
                s = wmma_bf16(Qa[c], lds_v16(kp + c * 32), s);
            S[n] = s;
        }
        // ---- causal mask (-10000 replace, in log2 units) + online softmax ----
#pragma unroll
        for (int j = 0; j < 8; ++j) {
            const int qrow = qbase + j + 8 * half;
#pragma unroll
            for (int n = 0; n < 4; ++n)
                if (kb + n * 16 + lm > qrow) S[n][j] = -14426.95f;
            float tmax = fmaxf(fmaxf(S[0][j], S[1][j]), fmaxf(S[2][j], S[3][j]));
#pragma unroll
            for (int m = 1; m < 16; m <<= 1) tmax = fmaxf(tmax, __shfl_xor(tmax, m, 32));
            float mnew = fmaxf(mrow[j], tmax);
            float scl = exp2f(mrow[j] - mnew);
            float p0 = exp2f(S[0][j] - mnew);
            float p1 = exp2f(S[1][j] - mnew);
            float p2 = exp2f(S[2][j] - mnew);
            float p3 = exp2f(S[3][j] - mnew);
            float rs = (p0 + p1) + (p2 + p3);
#pragma unroll
            for (int m = 1; m < 16; m <<= 1) rs += __shfl_xor(rs, m, 32);
            lsum[j] = lsum[j] * scl + rs;
            mrow[j] = mnew;
#pragma unroll
            for (int dt = 0; dt < 8; ++dt) acc[dt][j] *= scl;
            __bf16* pr = PLw + (j + 8 * half) * PROW + lm;
            pr[0]  = f2bf(p0);  pr[16] = f2bf(p1);
            pr[32] = f2bf(p2);  pr[48] = f2bf(p3);
        }
        asm volatile("s_wait_asynccnt 0" ::: "memory");   // V landed
        __syncthreads();

        // ---- PV: K-depth 64 = two A tiles from the wave-local P transpose ----
        v16bf pa0 = lds_v16(PLw + lm * PROW + half * 16);
        v16bf pa1 = lds_v16(PLw + lm * PROW + 32 + half * 16);
#pragma unroll
        for (int dt = 0; dt < 8; ++dt) {
            const __bf16* vp = VL + (dt * 16 + lm) * VROW + half * 16;
            acc[dt] = wmma_bf16(pa0, lds_v16(vp), acc[dt]);
            acc[dt] = wmma_bf16(pa1, lds_v16(vp + 32), acc[dt]);
        }
        __syncthreads();   // everyone done reading before next chunk overwrites
    }
#pragma unroll
    for (int dt = 0; dt < 8; ++dt)
#pragma unroll
        for (int j = 0; j < 8; ++j) {
            int tok = qbase + j + 8 * half;
            int col = h * HD + dt * 16 + lm;
            O[(size_t)tok * DM + col] = f2bf(acc[dt][j] / lsum[j]);
        }
}

// ---------------------------------------------------------------------------
// gemm: out[t][n] = sum_c O[t][c] * Wb[n][c]   (2048^3, bf16 -> f32)
// One wave per 16x128 C tile (8 accumulators), K step 32.
// ---------------------------------------------------------------------------
__global__ __launch_bounds__(256) void gemm_kernel(
    const __bf16* __restrict__ A, const __bf16* __restrict__ Wb,
    float* __restrict__ out)
{
    const int wave = threadIdx.x >> 5, lane = threadIdx.x & 31;
    const int half = lane >> 4, lm = lane & 15;
    const int task = blockIdx.x * 8 + wave;            // 2048 tasks
    const int mbase = (task >> 4) * 16;                // 128 M tiles
    const int nbase = (task & 15) * 128;               // 16 N groups
    v8f acc[8] = {};
    const __bf16* arow = A + (size_t)(mbase + lm) * DM + half * 16;
    for (int k0 = 0; k0 < DM; k0 += 32) {
        v16bf a = *(const v16bf*)(arow + k0);
#pragma unroll
        for (int nt = 0; nt < 8; ++nt) {
            const __bf16* bp = Wb + (size_t)(nbase + nt * 16 + lm) * DM + half * 16 + k0;
            acc[nt] = wmma_bf16(a, *(const v16bf*)bp, acc[nt]);
        }
    }
#pragma unroll
    for (int nt = 0; nt < 8; ++nt)
#pragma unroll
        for (int j = 0; j < 8; ++j)
            out[(size_t)(mbase + j + 8 * half) * DM + nbase + nt * 16 + lm] = acc[nt][j];
}

// ---------------------------------------------------------------------------
extern "C" void kernel_launch(void* const* d_in, const int* in_sizes, int n_in,
                              void* d_out, int out_size, void* d_ws, size_t ws_size,
                              hipStream_t stream) {
    (void)in_sizes; (void)n_in; (void)out_size; (void)ws_size;
    const float* q    = (const float*)d_in[0];
    const float* k    = (const float*)d_in[1];
    const float* v    = (const float*)d_in[2];
    // d_in[3] attention_mask: strictly-upper-triangular causal -> applied structurally
    const int*   pos  = (const int*)d_in[4];
    const float* qw   = (const float*)d_in[5];
    const float* kw   = (const float*)d_in[6];
    const float* wout = (const float*)d_in[7];
    float* out = (float*)d_out;

    char* ws = (char*)d_ws;
    const size_t MB = 1ull << 20;
    __bf16* Qh = (__bf16*)(ws);             //  8 MB  [16][2048][128]
    __bf16* Kh = (__bf16*)(ws +  8 * MB);   //  2 MB  [4][2048][128]
    __bf16* Vt = (__bf16*)(ws + 10 * MB);   //  2 MB  [4][128][2048]
    __bf16* Wb = (__bf16*)(ws + 12 * MB);   //  8 MB  [2048][2048]
    __bf16* O  = (__bf16*)(ws + 20 * MB);   //  8 MB  [2048][2048]

    prep_kernel<<<T_LEN, 256, 0, stream>>>(q, k, v, pos, qw, kw, Qh, Kh, Vt);
    convw_kernel<<<(DM * DM + 255) / 256, 256, 0, stream>>>(wout, Wb, DM * DM);
    attn_kernel<<<512, 128, 0, stream>>>(Qh, Kh, Vt, O);
    gemm_kernel<<<256, 256, 0, stream>>>(O, Wb, out);
}
